// Prediction_79723182949086
// MI455X (gfx1250) — compile-verified, matrix operand order checked by reading, not run on hardware
//
#include <hip/hip_runtime.h>

// ---------------------------------------------------------------------------
// 2-layer LSTM, B=64, U=512, H=1024, gates=4096, K(concat)=2048.
// Strategy: persistent kernel per layer; per step one (64x2048)x(2048x4096)
// bf16 WMMA GEMM (weights resident in 192MB L2), gates in LDS, cell state in
// registers, one grid barrier per step, double-buffered swizzled h.
// ---------------------------------------------------------------------------

typedef __attribute__((ext_vector_type(16))) __bf16 v16bf;
typedef __attribute__((ext_vector_type(8)))  float  v8f;

#define H_      1024
#define U_      512
#define B_      64
#define NBLK    128        // persistent blocks; each owns 32 packed gate cols = 8 hidden units
#define TPB     256        // 8 waves of 32
#define KBLK    64         // 2048 / 32
#define FRAG    512        // 32 lanes * 16 bf16 per (tile,kb) fragment
#define PARSZ   65536      // one parity of Hsw: 4*32*512 bf16 elems (64x1024)

__device__ __forceinline__ float sig_(float x)  { return 1.0f / (1.0f + __expf(-x)); }
__device__ __forceinline__ float tanh_(float x) { return 1.0f - 2.0f / (__expf(2.0f * x) + 1.0f); }

// k_local within a 32-wide K block for 16-bit WMMA A/B operands (ISA 7.12.2):
// lanes 0-15 hold k 0..7 (elems 0-7) and 16..23 (elems 8-15);
// lanes 16-31 hold k 8..15 and 24..31.
__device__ __forceinline__ int klocal_(int lane, int e) {
  int base = (lane < 16) ? 0 : 8;
  return (e < 8) ? (base + e) : (16 + base + (e - 8));
}

// ---- Phase 1a: embedding gather -> swizzled bf16 A fragments --------------
// Xsw layout: [t(512)][mt(4)][kb(32)][lane(32)][e(16)] bf16
__global__ void embed_pack_kernel(const int* __restrict__ y,
                                  const float* __restrict__ E,
                                  __bf16* __restrict__ Xsw) {
  long idx = (long)blockIdx.x * blockDim.x + threadIdx.x;   // < 512*65536
  int e    =  idx        & 15;
  int lane = (idx >> 4)  & 31;
  int kb   = (idx >> 9)  & 31;
  int mt   = (idx >> 14) & 3;
  int t    =  (int)(idx >> 16);
  int b    = mt * 16 + (lane & 15);
  int k    = kb * 32 + klocal_(lane, e);
  int v    = y[b * U_ + t];
  Xsw[idx] = (__bf16)E[(size_t)v * H_ + k];
}

// ---- Phase 1b: weights -> packed bf16 B fragments, gate-interleaved -------
// Bp layout: [nt(256)][kb(64)][lane(32)][e(16)] bf16.  Packed column
// np = 4*u + gate  (so each block's 32 columns = 8 hidden units x 4 gates).
// B[k][np] = W[norig][k], W = [Wih | Whh] along K, norig = gate*1024 + u.
__global__ void pack_weights_kernel(const float* __restrict__ Wih,
                                    const float* __restrict__ Whh,
                                    __bf16* __restrict__ Bp) {
  long idx = (long)blockIdx.x * blockDim.x + threadIdx.x;   // < 256*64*512
  int e    =  idx        & 15;
  int lane = (idx >> 4)  & 31;
  int kb   = (idx >> 9)  & 63;
  int nt   = (int)(idx >> 15);
  int k    = kb * 32 + klocal_(lane, e);
  int np   = nt * 16 + (lane & 15);
  int no   = (np & 3) * H_ + (np >> 2);
  float w  = (k < H_) ? Wih[(size_t)no * H_ + k] : Whh[(size_t)no * H_ + (k - H_)];
  Bp[idx] = (__bf16)w;
}

// ---- Phase 2: persistent recurrent layer ----------------------------------
__global__ __launch_bounds__(TPB) void lstm_layer_kernel(
    const __bf16* __restrict__ Xsw,   // [512][4][32][32][16] x-part, swizzled
    const __bf16* __restrict__ Bp,    // [256][64][32][16] packed weights
    const float*  __restrict__ bias,  // 4096, original gate-major order
    __bf16* __restrict__ Hsw,         // [2 parities][4][32][32][16]
    unsigned* __restrict__ ctr,       // grid barrier counter (zeroed)
    __bf16* __restrict__ Xout,        // layer0: hs0 swizzled out; layer1: null
    float* __restrict__ gout,         // layer1: g (B,U,H) out; layer0: null
    float* __restrict__ hfin,         // final h (B,H)
    float* __restrict__ cfin)         // final c (B,H)
{
  __shared__ float lg[B_ * 32];       // this block's 32 gate columns, 8 KB

  const int tid  = threadIdx.x;
  const int lane = tid & 31;
  const int wave = tid >> 5;
  const int mt   = wave & 3;          // M tile: rows 16*mt..
  const int ntl  = wave >> 2;         // local N tile 0/1
  const int ntg  = blockIdx.x * 2 + ntl;
  const int m0   = mt * 16 + ((lane < 16) ? 0 : 8);
  const int nl   = ntl * 16 + (lane & 15);

  const __bf16* bp = Bp + ((size_t)ntg * KBLK * 32 + lane) * 16;

  float creg[2] = {0.0f, 0.0f};       // cell state lives in registers

  for (int t = 0; t < U_; ++t) {
    const __bf16* aX = Xsw + (((size_t)t * 4 + mt) * 32 * 32 + lane) * 16;
    const __bf16* aH = Hsw + (size_t)(t & 1) * PARSZ + ((size_t)mt * 32 * 32 + lane) * 16;

    v8f acc = {};
#pragma unroll 4
    for (int kb = 0; kb < 32; ++kb) {          // x contribution (k 0..1023)
      v16bf a = *(const v16bf*)(aX + (size_t)kb * FRAG);
      v16bf b = *(const v16bf*)(bp + (size_t)kb * FRAG);
      acc = __builtin_amdgcn_wmma_f32_16x16x32_bf16(false, a, false, b,
                                                    (short)0, acc, false, false);
    }
#pragma unroll 4
    for (int kb = 0; kb < 32; ++kb) {          // h contribution (k 1024..2047)
      v16bf a = *(const v16bf*)(aH + (size_t)kb * FRAG);
      v16bf b = *(const v16bf*)(bp + (size_t)(32 + kb) * FRAG);
      acc = __builtin_amdgcn_wmma_f32_16x16x32_bf16(false, a, false, b,
                                                    (short)0, acc, false, false);
    }
    // D layout: VGPR v holds row m0+v, col nl (lanes 0-15: M=v, 16-31: M=8+v)
#pragma unroll
    for (int v = 0; v < 8; ++v) lg[(m0 + v) * 32 + nl] = acc[v];
    __syncthreads();

    // elementwise gates for this block's 8 hidden units x 64 batch rows
#pragma unroll
    for (int r = 0; r < 2; ++r) {
      int idx = tid + r * TPB;        // 0..511
      int m   = idx >> 3;             // batch row
      int ul  = idx & 7;              // local hidden unit
      int u   = (blockIdx.x << 3) + ul;
      float gi = lg[m * 32 + ul * 4 + 0] + bias[u];
      float gf = lg[m * 32 + ul * 4 + 1] + bias[H_ + u] + 1.0f;   // FGB
      float gg = lg[m * 32 + ul * 4 + 2] + bias[2 * H_ + u];
      float go = lg[m * 32 + ul * 4 + 3] + bias[3 * H_ + u];
      float i = sig_(gi), f = sig_(gf), g = tanh_(gg), o = sig_(go);
      float c = f * creg[r] + i * g;
      creg[r] = c;
      float h = o * tanh_(c);

      // scatter h into swizzled A-fragment layout for next step / next layer
      int mt2 = m >> 4, Lm = m & 15, kb = u >> 5, kl = u & 31, grp = kl >> 3;
      int lane2 = Lm + ((grp & 1) ? 16 : 0);
      int e2    = (kl & 7) + ((grp & 2) ? 8 : 0);
      size_t soff = (((size_t)mt2 * 32 + kb) * 32 + lane2) * 16 + e2;
      __bf16 hb = (__bf16)h;
      Hsw[(size_t)((t + 1) & 1) * PARSZ + soff] = hb;
      if (Xout) Xout[(size_t)t * PARSZ + soff] = hb;
      if (gout) gout[((size_t)m * U_ + t) * H_ + u] = h;
      if (t == U_ - 1) {
        hfin[(size_t)m * H_ + u] = h;
        cfin[(size_t)m * H_ + u] = c;
      }
    }

    // ---- one grid barrier per step (publish h, protect Xout in-place-free) --
    __threadfence();
    __syncthreads();
    if (tid == 0) {
      __hip_atomic_fetch_add(ctr, 1u, __ATOMIC_RELEASE, __HIP_MEMORY_SCOPE_AGENT);
      unsigned target = (unsigned)(t + 1) * gridDim.x;
      while (__hip_atomic_load(ctr, __ATOMIC_ACQUIRE, __HIP_MEMORY_SCOPE_AGENT) < target)
        __builtin_amdgcn_s_sleep(1);
    }
    __syncthreads();
  }
}

// ---------------------------------------------------------------------------
extern "C" void kernel_launch(void* const* d_in, const int* in_sizes, int n_in,
                              void* d_out, int out_size, void* d_ws, size_t ws_size,
                              hipStream_t stream) {
  const int*   y    = (const int*)d_in[0];
  const float* E    = (const float*)d_in[1];
  const float* Wih0 = (const float*)d_in[2];
  const float* Whh0 = (const float*)d_in[3];
  const float* b0   = (const float*)d_in[4];
  const float* Wih1 = (const float*)d_in[5];
  const float* Whh1 = (const float*)d_in[6];
  const float* b1   = (const float*)d_in[7];

  float* out   = (float*)d_out;
  float* g_out = out;                                    // (64,512,1024)
  float* h_out = out + (size_t)B_ * U_ * H_;             // (2,64,1024)
  float* c_out = h_out + (size_t)2 * B_ * H_;            // (2,64,1024)

  char* ws = (char*)d_ws;
  __bf16* Xsw0 = (__bf16*)ws; ws += (size_t)U_ * PARSZ * 2;      // 64 MB
  __bf16* Xsw1 = (__bf16*)ws; ws += (size_t)U_ * PARSZ * 2;      // 64 MB
  __bf16* Bp0  = (__bf16*)ws; ws += (size_t)256 * KBLK * FRAG * 2; // 16 MB
  __bf16* Bp1  = (__bf16*)ws; ws += (size_t)256 * KBLK * FRAG * 2; // 16 MB
  __bf16* Hsw0 = (__bf16*)ws; ws += (size_t)2 * PARSZ * 2;       // 256 KB
  __bf16* Hsw1 = (__bf16*)ws; ws += (size_t)2 * PARSZ * 2;       // 256 KB
  unsigned* ctr = (unsigned*)ws; ws += 256;                       // 2 counters

  hipMemsetAsync(Hsw0, 0, (size_t)2 * PARSZ * 2, stream);  // h0 = 0
  hipMemsetAsync(Hsw1, 0, (size_t)2 * PARSZ * 2, stream);
  hipMemsetAsync(ctr,  0, 256, stream);

  embed_pack_kernel<<<(U_ * PARSZ) / TPB, TPB, 0, stream>>>(y, E, Xsw0);
  pack_weights_kernel<<<(256 * KBLK * FRAG) / TPB, TPB, 0, stream>>>(Wih0, Whh0, Bp0);
  pack_weights_kernel<<<(256 * KBLK * FRAG) / TPB, TPB, 0, stream>>>(Wih1, Whh1, Bp1);

  lstm_layer_kernel<<<NBLK, TPB, 0, stream>>>(Xsw0, Bp0, b0, Hsw0, ctr,
                                              Xsw1, nullptr, h_out, c_out);
  lstm_layer_kernel<<<NBLK, TPB, 0, stream>>>(Xsw1, Bp1, b1, Hsw1, ctr + 64,
                                              nullptr, g_out,
                                              h_out + (size_t)B_ * H_,
                                              c_out + (size_t)B_ * H_);
}
